// AttentionModel_57380763075118
// MI455X (gfx1250) — matrix operand order repaired and checked by reading, not compiled, not run on hardware
//
#include <hip/hip_runtime.h>
#include <math.h>

// ---------------- problem constants ----------------
#define NFFT   2048
#define HOP    512
#define BINS   1025            // NFFT/2+1
#define HEADS  2
#define HD     1025            // D/HEADS
#define DMODEL 2050            // 2*BINS
#define DFF    8200            // 4*D
#define BATCH  4
#define LEN    441000
#define TFR    862             // 1 + LEN/HOP with reflect pad NFFT/2
#define PADL   1024
#define NSIG   8               // BATCH*2 channels
#define OUTLEN (NFFT + HOP*(TFR-1))
#define ESTIM_N ((long)NSIG*(long)LEN)

// ---------------- CDNA5 types ----------------
typedef __attribute__((ext_vector_type(16))) __bf16 v16bf;
typedef __attribute__((ext_vector_type(8)))  float  v8f;
typedef unsigned int u32x4v __attribute__((ext_vector_type(4)));
typedef int          i32x8v __attribute__((ext_vector_type(8)));
typedef int          i32x4v __attribute__((ext_vector_type(4)));

union FragBF { v16bf v; unsigned short u[16]; u32x4v q[2]; };
union AccF   { v8f   v; float f[8]; };

#if defined(__has_builtin)
#  if __has_builtin(__builtin_amdgcn_tensor_load_to_lds) && __has_builtin(__builtin_amdgcn_s_wait_tensorcnt)
#    define HAVE_TDM 1
#  endif
#endif
#ifndef HAVE_TDM
#  define HAVE_TDM 0
#endif

__device__ __forceinline__ unsigned short f2bf(float f) {
  unsigned int u = __float_as_uint(f);
  u += 0x7fffu + ((u >> 16) & 1u);          // round-to-nearest-even
  return (unsigned short)(u >> 16);
}
__device__ __forceinline__ float bf2f(unsigned short h) {
  return __uint_as_float(((unsigned int)h) << 16);
}

#if HAVE_TDM
// Build a D# (ISA ch.8) for a 2D tile of 16-bit elements and issue
// TENSOR_LOAD_TO_LDS. tensor dims == tile dims (tile is known in-bounds,
// so the TDM OOB path is never exercised). Row stride in elements.
// This toolchain's builtin takes 6 args: (g0 x4, g1 x8, g2 x4, g3 x4, x8, cpol);
// groups 2/3 and the trailing group are zero for a 2D tile.
__device__ __forceinline__ void tdm_load_tile_2d(
    unsigned lds_off, const void* gptr, unsigned long long stride_elems,
    unsigned tile0 /*row length*/, unsigned tile1 /*rows*/)
{
  unsigned long long ga = (unsigned long long)gptr;
  u32x4v g0;
  g0[0] = 1u;                                        // count=1, user D#
  g0[1] = lds_off;                                   // lds_addr (bytes)
  g0[2] = (unsigned)(ga & 0xFFFFFFFFu);              // global_addr[31:0]
  g0[3] = (unsigned)((ga >> 32) & 0x01FFFFFFu) | (2u << 30);  // addr[56:32] | type=2
  i32x8v g1;
  g1[0] = (int)(1u << 16);                           // data_size=1 (2 bytes)
  g1[1] = (int)((tile0 & 0xFFFFu) << 16);            // tensor_dim0[15:0]
  g1[2] = (int)(((tile0 >> 16) & 0xFFFFu) | ((tile1 & 0xFFFFu) << 16)); // tdim0 hi | tdim1 lo
  g1[3] = (int)(((tile1 >> 16) & 0xFFFFu) | ((tile0 & 0xFFFFu) << 16)); // tdim1 hi | tile_dim0
  g1[4] = (int)(tile1 & 0xFFFFu);                    // tile_dim1 | tile_dim2=0
  g1[5] = (int)(unsigned)(stride_elems & 0xFFFFFFFFu);        // tensor_dim0_stride lo
  g1[6] = (int)(unsigned)((stride_elems >> 32) & 0xFFFFu);    // stride hi | dim1_stride lo=0
  g1[7] = 0;
  i32x4v gz = {0, 0, 0, 0};
  i32x8v z8 = {0, 0, 0, 0, 0, 0, 0, 0};
  __builtin_amdgcn_tensor_load_to_lds(g0, g1, gz, gz, z8, 0);
}
#endif

// =====================================================================
// Strided-batched bf16 GEMM:  C += alpha * A @ op(B)
//   BT=true : B stored row-major (N,K)  ->  C = A @ B^T
//   BT=false: B stored row-major (K,N)  ->  C = A @ B
// One wave computes a 32x32 tile. Interior tiles are DMA'd to LDS by the
// Tensor Data Mover (TENSOR_LOAD_TO_LDS + s_wait_tensorcnt); edge tiles use
// a guarded cooperative fill. Fragments are gathered from LDS with
// ds_load_b128 in contiguous k order (k = 16*half + s, same for A and B), then v_wmma_f32_16x16x32_bf16.
// blockIdx.z = b*Hn + h with independent (b,h) strides for A/B/C.
// =====================================================================
template <bool BT>
__global__ __launch_bounds__(32) void gemm_wmma(
    const unsigned short* __restrict__ A, const unsigned short* __restrict__ Bm,
    float* __restrict__ C,
    int M, int N, int K, long lda, long ldb, long ldc,
    long sAb, long sAh, long sBb, long sBh, long sCb, long sCh,
    int Hn, float alpha)
{
  __shared__ __align__(16) unsigned short tileA[32 * 32];
  __shared__ __align__(16) unsigned short tileB[32 * 32];

  int z = blockIdx.z; int bb = z / Hn, hh = z % Hn;
  A  += (long)bb * sAb + (long)hh * sAh;
  Bm += (long)bb * sBb + (long)hh * sBh;
  C  += (long)bb * sCb + (long)hh * sCh;

  const int lane   = threadIdx.x;
  const int lane16 = lane & 15;
  const int upper  = (lane >= 16) ? 1 : 0;
  const int m0 = blockIdx.y * 32;
  const int n0 = blockIdx.x * 32;
  const bool mFull = (m0 + 32 <= M);
  const bool nFull = (n0 + 32 <= N);

#if HAVE_TDM
  const unsigned ldsA = (unsigned)(unsigned long long)(void*)tileA;
  const unsigned ldsB = (unsigned)(unsigned long long)(void*)tileB;
#endif

  AccF acc[2][2];
#pragma unroll
  for (int i = 0; i < 2; ++i)
#pragma unroll
    for (int j = 0; j < 2; ++j)
#pragma unroll
      for (int v = 0; v < 8; ++v) acc[i][j].f[v] = 0.0f;

  for (int k0 = 0; k0 < K; k0 += 32) {
    const bool kFull = (k0 + 32 <= K);
    bool fa = mFull && kFull;
    bool fb = nFull && kFull;
#if HAVE_TDM
    // previous iteration's LDS gathers are consumed by the WMMAs above;
    // defensive drain before the TDM overwrites the tiles.
    __asm__ volatile("s_wait_dscnt 0x0" ::: "memory");
    if (fa) tdm_load_tile_2d(ldsA, A + (long)m0 * lda + k0, (unsigned long long)lda, 32u, 32u);
    if (fb) tdm_load_tile_2d(ldsB,
                             BT ? (Bm + (long)n0 * ldb + k0) : (Bm + (long)k0 * ldb + n0),
                             (unsigned long long)ldb, 32u, 32u);
#else
    fa = false; fb = false;
#endif
    if (!fa) {  // guarded cooperative fill (edge tiles / no-TDM build)
      for (int idx = lane; idx < 1024; idx += 32) {
        int r = idx >> 5, c = idx & 31;
        int row = m0 + r, k = k0 + c;
        unsigned short v = 0;
        if (row < M && k < K) v = A[(long)row * lda + k];
        tileA[idx] = v;
      }
    }
    if (!fb) {
      for (int idx = lane; idx < 1024; idx += 32) {
        int r = idx >> 5, c = idx & 31;
        unsigned short v = 0;
        if (BT) {
          int col = n0 + r, k = k0 + c;
          if (col < N && k < K) v = Bm[(long)col * ldb + k];
        } else {
          int k = k0 + r, col = n0 + c;
          if (k < K && col < N) v = Bm[(long)k * ldb + col];
        }
        tileB[idx] = v;
      }
    }
#if HAVE_TDM
    if (fa || fb) __builtin_amdgcn_s_wait_tensorcnt(0);
#endif
    __asm__ volatile("" ::: "memory");   // keep LDS gathers below the fills

    // ---- fragment gathers from LDS (contiguous k = 16*half + s for A and B) ----
    FragBF af[2], bf[2];
    const u32x4v* lA = (const u32x4v*)tileA;
#pragma unroll
    for (int sub = 0; sub < 2; ++sub) {
      int r = sub * 16 + lane16;                  // A row within tile
      af[sub].q[0] = lA[r * 4 + upper * 2];       // K = upper*16 + 0..7
      af[sub].q[1] = lA[r * 4 + upper * 2 + 1];   // K = upper*16 + 8..15
    }
    if (BT) {
      const u32x4v* lB = (const u32x4v*)tileB;
#pragma unroll
      for (int sub = 0; sub < 2; ++sub) {
        int r = sub * 16 + lane16;                // B col (tile stored N x K)
        bf[sub].q[0] = lB[r * 4 + upper * 2];     // K = upper*16 + 0..7
        bf[sub].q[1] = lB[r * 4 + upper * 2 + 1]; // K = upper*16 + 8..15
      }
    } else {
#pragma unroll
      for (int sub = 0; sub < 2; ++sub) {
        int cc = sub * 16 + lane16;               // tile stored K x N: strided gather
#pragma unroll
        for (int e = 0; e < 16; ++e)
          bf[sub].u[e] = tileB[(e + upper * 16) * 32 + cc];
      }
    }

#pragma unroll
    for (int i = 0; i < 2; ++i)
#pragma unroll
      for (int j = 0; j < 2; ++j)
        acc[i][j].v = __builtin_amdgcn_wmma_f32_16x16x32_bf16(
            false, af[i].v, false, bf[j].v, (short)0, acc[i][j].v, false, false);
  }

#pragma unroll
  for (int i = 0; i < 2; ++i)
#pragma unroll
    for (int j = 0; j < 2; ++j) {
      int col = n0 + j * 16 + lane16;
      if (col >= N) continue;
#pragma unroll
      for (int v = 0; v < 8; ++v) {
        int row = m0 + i * 16 + v + upper * 8;    // ISA f32 C/D layout
        if (row < M) C[(long)row * ldc + col] += alpha * acc[i][j].f[v];
      }
    }
}

// =====================================================================
// Elementwise / reduction kernels
// =====================================================================
__global__ void rowbias_init_k(float* __restrict__ C, const float* b1, float s1,
                               const float* b2, float s2, long n, int ncol) {
  long i = (long)blockIdx.x * blockDim.x + threadIdx.x;
  if (i >= n) return;
  int c = (int)(i % ncol);
  float v = 0.0f;
  if (b1) v += s1 * b1[c];
  if (b2) v += s2 * b2[c];
  C[i] = v;
}

__global__ void f32_to_bf16_k(const float* __restrict__ in, unsigned short* __restrict__ out, long n) {
  long i = (long)blockIdx.x * blockDim.x + threadIdx.x;
  if (i < n) out[i] = f2bf(in[i]);
}

__global__ void relu_bf16_k(const float* __restrict__ in, unsigned short* __restrict__ out, long n) {
  long i = (long)blockIdx.x * blockDim.x + threadIdx.x;
  if (i < n) out[i] = f2bf(fmaxf(in[i], 0.0f));
}

__global__ void build_stft_mats_k(unsigned short* __restrict__ cosW, unsigned short* __restrict__ sinW) {
  long i = (long)blockIdx.x * blockDim.x + threadIdx.x;
  if (i >= (long)BINS * NFFT) return;
  int n = (int)(i / NFFT), k = (int)(i % NFFT);
  long kn = ((long)n * k) % NFFT;
  double ang = 6.283185307179586476925286766559 * (double)kn / (double)NFFT;
  cosW[i] = f2bf((float)cos(ang));
  sinW[i] = f2bf((float)(-sin(ang)));
}

__global__ void build_istft_mats_k(unsigned short* __restrict__ icosW, unsigned short* __restrict__ isinW) {
  long i = (long)blockIdx.x * blockDim.x + threadIdx.x;
  if (i >= (long)NFFT * BINS) return;
  int t = (int)(i / BINS), n = (int)(i % BINS);
  long tn = ((long)t * n) % NFFT;
  double ang = 6.283185307179586476925286766559 * (double)tn / (double)NFFT;
  float coef = (n == 0 || n == BINS - 1) ? 1.0f : 2.0f;
  icosW[i] = f2bf((float)( coef * cos(ang) / (double)NFFT));
  isinW[i] = f2bf((float)(-coef * sin(ang) / (double)NFFT));
}

__global__ void frame_window_k(const float* __restrict__ mix, const float* __restrict__ win,
                               unsigned short* __restrict__ frames) {
  long i = (long)blockIdx.x * blockDim.x + threadIdx.x;
  long n = (long)NSIG * TFR * NFFT;
  if (i >= n) return;
  int k = (int)(i % NFFT);
  long r = i / NFFT;
  int t = (int)(r % TFR);
  int s = (int)(r / TFR);
  long p = (long)t * HOP + k - PADL;
  if (p < 0) p = -p;                       // reflect pad
  if (p >= LEN) p = 2L * LEN - 2 - p;
  frames[i] = f2bf(mix[(long)s * LEN + p] * win[k]);
}

__global__ void conv1_posenc_pack_k(const float* __restrict__ specR, const float* __restrict__ specI,
                                    const float* c1wr, const float* c1br, const float* c1wi, const float* c1bi,
                                    unsigned short* __restrict__ nrb, unsigned short* __restrict__ nib) {
  long i = (long)blockIdx.x * blockDim.x + threadIdx.x;
  long n = (long)BATCH * TFR * DMODEL;
  if (i >= n) return;
  int d = (int)(i % DMODEL);
  long r = i / DMODEL;
  int t = (int)(r % TFR);
  int b = (int)(r / TFR);
  int bin = d >> 1, c = d & 1;
  long sidx = ((long)(b * 2 + c) * TFR + t) * BINS + bin;   // spec stored (sig,T,BINS)
  float re = specR[sidx], im = specI[sidx];
  float wr = c1wr[c], br = c1br[c], wi = c1wi[c], bi = c1bi[c];
  float nr = (re * wr + br) - (im * wi + bi);
  float ni = (im * wr + br) + (re * wi + bi);
  float ang = (float)t * expf(-((float)(d & ~1)) * (logf(10000.0f) / (float)DMODEL));
  float pe = (d & 1) ? cosf(ang) : sinf(ang);
  nrb[i] = f2bf(nr + pe);
  nib[i] = f2bf(ni + pe);
}

__global__ void softmax_rows_k(const float* __restrict__ in, unsigned short* __restrict__ out,
                               int ncol, float scale) {
  __shared__ float red[256];
  int r = blockIdx.x;
  const float* x = in + (long)r * ncol;
  float m = -1e30f;
  for (int c = threadIdx.x; c < ncol; c += blockDim.x) m = fmaxf(m, x[c] * scale);
  red[threadIdx.x] = m; __syncthreads();
  for (int s = 128; s > 0; s >>= 1) {
    if ((int)threadIdx.x < s) red[threadIdx.x] = fmaxf(red[threadIdx.x], red[threadIdx.x + s]);
    __syncthreads();
  }
  m = red[0]; __syncthreads();
  float sum = 0.0f;
  for (int c = threadIdx.x; c < ncol; c += blockDim.x) sum += expf(x[c] * scale - m);
  red[threadIdx.x] = sum; __syncthreads();
  for (int s = 128; s > 0; s >>= 1) {
    if ((int)threadIdx.x < s) red[threadIdx.x] += red[threadIdx.x + s];
    __syncthreads();
  }
  float inv = 1.0f / red[0];
  for (int c = threadIdx.x; c < ncol; c += blockDim.x)
    out[(long)r * ncol + c] = f2bf(expf(x[c] * scale - m) * inv);
}

__global__ void layernorm_rows_k(const float* __restrict__ in, const float* g, const float* bb,
                                 float* outF, unsigned short* outB, int ncol) {
  __shared__ float r1[256];
  __shared__ float r2[256];
  int r = blockIdx.x;
  const float* x = in + (long)r * ncol;
  float s = 0.0f, sq = 0.0f;
  for (int c = threadIdx.x; c < ncol; c += blockDim.x) { float v = x[c]; s += v; sq += v * v; }
  r1[threadIdx.x] = s; r2[threadIdx.x] = sq; __syncthreads();
  for (int st = 128; st > 0; st >>= 1) {
    if ((int)threadIdx.x < st) { r1[threadIdx.x] += r1[threadIdx.x + st]; r2[threadIdx.x] += r2[threadIdx.x + st]; }
    __syncthreads();
  }
  float mean = r1[0] / (float)ncol;
  float var  = r2[0] / (float)ncol - mean * mean;
  float rstd = rsqrtf(var + 1e-5f);
  for (int c = threadIdx.x; c < ncol; c += blockDim.x) {
    float y = (x[c] - mean) * rstd * g[c] + bb[c];
    if (outF) outF[(long)r * ncol + c] = y;
    if (outB) outB[(long)r * ncol + c] = f2bf(y);
  }
}

__global__ void merge_heads_k(const float* __restrict__ src, unsigned short* __restrict__ dst) {
  long i = (long)blockIdx.x * blockDim.x + threadIdx.x;
  long n = (long)BATCH * TFR * DMODEL;
  if (i >= n) return;
  int d = (int)(i % DMODEL);
  long r = i / DMODEL;
  int s = (int)(r % TFR);
  int b = (int)(r / TFR);
  int h = d / HD, dh = d % HD;
  dst[i] = f2bf(src[(((long)b * HEADS + h) * TFR + s) * HD + dh]);
}

__global__ void mask_apply_k(const float* __restrict__ fr, const float* __restrict__ fi,
                             const float* __restrict__ specR, const float* __restrict__ specI,
                             const float* c2wr, const float* c2br, const float* c2wi, const float* c2bi,
                             unsigned short* __restrict__ s2Rb, unsigned short* __restrict__ s2Ib,
                             float* __restrict__ outSpec) {
  long i = (long)blockIdx.x * blockDim.x + threadIdx.x;
  long n = (long)NSIG * TFR * BINS;
  if (i >= n) return;
  int bin = (int)(i % BINS);
  long r = i / BINS;
  int t = (int)(r % TFR);
  int sig = (int)(r / TFR);
  int b = sig >> 1, c = sig & 1;
  int d = bin * 2 + c;
  float xr = fr[((long)b * TFR + t) * DMODEL + d];
  float xi = fi[((long)b * TFR + t) * DMODEL + d];
  float mr = (xr * c2wr[c] + c2br[c]) - (xi * c2wi[c] + c2bi[c]);
  float mi = (xi * c2wr[c] + c2br[c]) + (xr * c2wi[c] + c2bi[c]);
  float rr = specR[i] * (1.0f / (1.0f + expf(-mr)));
  float ii = specI[i] * (1.0f / (1.0f + expf(-mi)));
  s2Rb[i] = f2bf(rr);
  s2Ib[i] = f2bf(ii);
  long o = (((long)sig * BINS + bin) * TFR + t) * 2;      // (8,BINS,T,2)
  outSpec[o] = rr; outSpec[o + 1] = ii;
}

__global__ void wsq_k(const float* __restrict__ win, float* __restrict__ wsq) {
  long i = (long)blockIdx.x * blockDim.x + threadIdx.x;
  long n = (long)TFR * NFFT;
  if (i >= n) return;
  int k = (int)(i % NFFT);
  int t = (int)(i / NFFT);
  atomicAdd(&wsq[(long)t * HOP + k], win[k] * win[k]);
}

__global__ void ola_k(const float* __restrict__ frames, const float* __restrict__ win,
                      float* __restrict__ outAcc) {
  long i = (long)blockIdx.x * blockDim.x + threadIdx.x;
  long n = (long)NSIG * TFR * NFFT;
  if (i >= n) return;
  int k = (int)(i % NFFT);
  long r = i / NFFT;
  int t = (int)(r % TFR);
  int s = (int)(r / TFR);
  atomicAdd(&outAcc[(long)s * OUTLEN + (long)t * HOP + k], frames[i] * win[k]);
}

__global__ void finalize_k(const float* __restrict__ outAcc, const float* __restrict__ wsq,
                           float* __restrict__ estim) {
  long i = (long)blockIdx.x * blockDim.x + threadIdx.x;
  if (i >= ESTIM_N) return;
  long j = i % LEN;
  int s = (int)(i / LEN);
  float w = wsq[PADL + j];
  float den = (w > 1e-11f) ? w : 1.0f;
  estim[i] = outAcc[(long)s * OUTLEN + PADL + j] / den;
}

// =====================================================================
// Host orchestration
// =====================================================================
extern "C" void kernel_launch(void* const* d_in, const int* in_sizes, int n_in,
                              void* d_out, int out_size, void* d_ws, size_t ws_size,
                              hipStream_t stream)
{
  (void)in_sizes; (void)n_in; (void)out_size; (void)ws_size;
  const float* mix    = (const float*)d_in[0];
  const float* window = (const float*)d_in[1];
  const float* c1wr = (const float*)d_in[2],  *c1br = (const float*)d_in[3];
  const float* c1wi = (const float*)d_in[4],  *c1bi = (const float*)d_in[5];
  const float* c2wr = (const float*)d_in[6],  *c2br = (const float*)d_in[7];
  const float* c2wi = (const float*)d_in[8],  *c2bi = (const float*)d_in[9];
  const float* Winw = (const float*)d_in[10], *binb = (const float*)d_in[11];
  const float* Wow  = (const float*)d_in[12], *bo   = (const float*)d_in[13];
  const float* l1wr = (const float*)d_in[14], *l1br = (const float*)d_in[15];
  const float* l1wi = (const float*)d_in[16], *l1bi = (const float*)d_in[17];
  const float* l2wr = (const float*)d_in[18], *l2br = (const float*)d_in[19];
  const float* l2wi = (const float*)d_in[20], *l2bi = (const float*)d_in[21];
  const float* n1gr = (const float*)d_in[22], *n1br = (const float*)d_in[23];
  const float* n1gi = (const float*)d_in[24], *n1bi = (const float*)d_in[25];
  const float* n2gr = (const float*)d_in[26], *n2br = (const float*)d_in[27];
  const float* n2gi = (const float*)d_in[28], *n2bi = (const float*)d_in[29];

  // ---- workspace bump allocator (~1.3 GB total) ----
  size_t off = 0;
  auto alloc = [&](size_t bytes) -> void* {
    void* p = (char*)d_ws + off;
    off += (bytes + 255) & ~(size_t)255;
    return p;
  };
  const long MROWS = (long)BATCH * TFR;        // 3448
  const long SROWS = (long)NSIG  * TFR;        // 6896

  unsigned short* cosW   = (unsigned short*)alloc((size_t)BINS * NFFT * 2);
  unsigned short* sinW   = (unsigned short*)alloc((size_t)BINS * NFFT * 2);
  unsigned short* icosW  = (unsigned short*)alloc((size_t)NFFT * BINS * 2);
  unsigned short* isinW  = (unsigned short*)alloc((size_t)NFFT * BINS * 2);
  unsigned short* framesB= (unsigned short*)alloc((size_t)SROWS * NFFT * 2);
  float* specR = (float*)alloc((size_t)SROWS * BINS * 4);
  float* specI = (float*)alloc((size_t)SROWS * BINS * 4);
  unsigned short* nrb = (unsigned short*)alloc((size_t)MROWS * DMODEL * 2);
  unsigned short* nib = (unsigned short*)alloc((size_t)MROWS * DMODEL * 2);
  unsigned short* WinB = (unsigned short*)alloc((size_t)3 * DMODEL * DMODEL * 2);
  unsigned short* WoB  = (unsigned short*)alloc((size_t)DMODEL * DMODEL * 2);
  unsigned short* l1wrB= (unsigned short*)alloc((size_t)DFF * DMODEL * 2);
  unsigned short* l1wiB= (unsigned short*)alloc((size_t)DFF * DMODEL * 2);
  unsigned short* l2wrB= (unsigned short*)alloc((size_t)DMODEL * DFF * 2);
  unsigned short* l2wiB= (unsigned short*)alloc((size_t)DMODEL * DFF * 2);
  float* qkvr = (float*)alloc((size_t)MROWS * 3 * DMODEL * 4);
  float* qkvi = (float*)alloc((size_t)MROWS * 3 * DMODEL * 4);
  unsigned short* qkvrb = (unsigned short*)alloc((size_t)MROWS * 3 * DMODEL * 2);
  unsigned short* qkvib = (unsigned short*)alloc((size_t)MROWS * 3 * DMODEL * 2);
  float* scoreF = (float*)alloc((size_t)NSIG * TFR * TFR * 4);
  unsigned short* probs[4];
  for (int m = 0; m < 4; ++m) probs[m] = (unsigned short*)alloc((size_t)NSIG * TFR * TFR * 2);
  float* outHr = (float*)alloc((size_t)NSIG * TFR * HD * 4);
  float* outHi = (float*)alloc((size_t)NSIG * TFR * HD * 4);
  unsigned short* mergeRb = (unsigned short*)alloc((size_t)MROWS * DMODEL * 2);
  unsigned short* mergeIb = (unsigned short*)alloc((size_t)MROWS * DMODEL * 2);
  float* attR = (float*)alloc((size_t)MROWS * DMODEL * 4);
  float* attI = (float*)alloc((size_t)MROWS * DMODEL * 4);
  unsigned short* normRb = (unsigned short*)alloc((size_t)MROWS * DMODEL * 2);
  unsigned short* normIb = (unsigned short*)alloc((size_t)MROWS * DMODEL * 2);
  float* hrF = (float*)alloc((size_t)MROWS * DFF * 4);
  float* hiF = (float*)alloc((size_t)MROWS * DFF * 4);
  unsigned short* hrB = (unsigned short*)alloc((size_t)MROWS * DFF * 2);
  unsigned short* hiB = (unsigned short*)alloc((size_t)MROWS * DFF * 2);
  float* l2rF = (float*)alloc((size_t)MROWS * DMODEL * 4);
  float* l2iF = (float*)alloc((size_t)MROWS * DMODEL * 4);
  float* frF  = (float*)alloc((size_t)MROWS * DMODEL * 4);
  float* fiF  = (float*)alloc((size_t)MROWS * DMODEL * 4);
  unsigned short* s2Rb = (unsigned short*)alloc((size_t)SROWS * BINS * 2);
  unsigned short* s2Ib = (unsigned short*)alloc((size_t)SROWS * BINS * 2);
  float* framesOut = (float*)alloc((size_t)SROWS * NFFT * 4);
  float* outAcc = (float*)alloc((size_t)NSIG * OUTLEN * 4);
  float* wsqBuf = (float*)alloc((size_t)OUTLEN * 4);

  float* estimOut = (float*)d_out;
  float* specOut  = (float*)d_out + ESTIM_N;

  dim3 TB(256);
  auto blks = [](long n) { return dim3((unsigned int)((n + 255) / 256)); };
  auto zero = [&](float* p, long n) {
    rowbias_init_k<<<blks(n), TB, 0, stream>>>(p, nullptr, 0.f, nullptr, 0.f, n, 1);
  };
  auto binit = [&](float* p, const float* b1, float s1, const float* b2, float s2, long rows, int ncol) {
    rowbias_init_k<<<blks(rows * ncol), TB, 0, stream>>>(p, b1, s1, b2, s2, rows * (long)ncol, ncol);
  };
  auto tobf = [&](const float* in, unsigned short* out, long n) {
    f32_to_bf16_k<<<blks(n), TB, 0, stream>>>(in, out, n);
  };
  auto gemm = [&](bool bt, const unsigned short* A, const unsigned short* Bm, float* C,
                  int M, int N, int K, long lda, long ldb, long ldc,
                  long sAb, long sAh, long sBb, long sBh, long sCb, long sCh,
                  int Hn, int batch, float alpha) {
    dim3 g((N + 31) / 32, (M + 31) / 32, batch), blk(32);
    if (bt) gemm_wmma<true ><<<g, blk, 0, stream>>>(A, Bm, C, M, N, K, lda, ldb, ldc,
                                                    sAb, sAh, sBb, sBh, sCb, sCh, Hn, alpha);
    else    gemm_wmma<false><<<g, blk, 0, stream>>>(A, Bm, C, M, N, K, lda, ldb, ldc,
                                                    sAb, sAh, sBb, sBh, sCb, sCh, Hn, alpha);
  };

  // ---- 0. weights -> bf16, DFT matrices ----
  tobf(Winw, WinB, (long)3 * DMODEL * DMODEL);
  tobf(Wow,  WoB,  (long)DMODEL * DMODEL);
  tobf(l1wr, l1wrB, (long)DFF * DMODEL);
  tobf(l1wi, l1wiB, (long)DFF * DMODEL);
  tobf(l2wr, l2wrB, (long)DMODEL * DFF);
  tobf(l2wi, l2wiB, (long)DMODEL * DFF);
  build_stft_mats_k<<<blks((long)BINS * NFFT), TB, 0, stream>>>(cosW, sinW);
  build_istft_mats_k<<<blks((long)NFFT * BINS), TB, 0, stream>>>(icosW, isinW);

  // ---- 1. STFT as DFT GEMM ----
  frame_window_k<<<blks(SROWS * NFFT), TB, 0, stream>>>(mix, window, framesB);
  zero(specR, SROWS * BINS); zero(specI, SROWS * BINS);
  gemm(true, framesB, cosW, specR, (int)SROWS, BINS, NFFT, NFFT, NFFT, BINS, 0,0,0,0,0,0, 1, 1, 1.0f);
  gemm(true, framesB, sinW, specI, (int)SROWS, BINS, NFFT, NFFT, NFFT, BINS, 0,0,0,0,0,0, 1, 1, 1.0f);

  // ---- 2. conv1x1 + positional encoding, pack (B,T,D) bf16 ----
  conv1_posenc_pack_k<<<blks(MROWS * DMODEL), TB, 0, stream>>>(specR, specI,
      c1wr, c1br, c1wi, c1bi, nrb, nib);

  // ---- 3. QKV projection (+bias) ----
  binit(qkvr, binb, 1.f, nullptr, 0.f, MROWS, 3 * DMODEL);
  binit(qkvi, binb, 1.f, nullptr, 0.f, MROWS, 3 * DMODEL);
  gemm(true, nrb, WinB, qkvr, (int)MROWS, 3 * DMODEL, DMODEL, DMODEL, DMODEL, 3 * DMODEL, 0,0,0,0,0,0, 1, 1, 1.0f);
  gemm(true, nib, WinB, qkvi, (int)MROWS, 3 * DMODEL, DMODEL, DMODEL, DMODEL, 3 * DMODEL, 0,0,0,0,0,0, 1, 1, 1.0f);
  tobf(qkvr, qkvrb, MROWS * 3 * DMODEL);
  tobf(qkvi, qkvib, MROWS * 3 * DMODEL);

  // ---- 4. four attention maps: scores -> softmax(bf16) ----
  const float scale = 1.0f / sqrtf((float)HD);
  const unsigned short* qsrc[4] = { qkvrb, qkvrb, qkvib, qkvib };   // rr, ri, ir, ii
  const unsigned short* ksrc[4] = { qkvrb, qkvib, qkvrb, qkvib };
  const long sQKb = (long)TFR * 3 * DMODEL;
  for (int m = 0; m < 4; ++m) {
    zero(scoreF, (long)NSIG * TFR * TFR);
    gemm(true, qsrc[m], ksrc[m] + DMODEL, scoreF, TFR, TFR, HD,
         3 * DMODEL, 3 * DMODEL, TFR,
         sQKb, HD, sQKb, HD, (long)HEADS * TFR * TFR, (long)TFR * TFR,
         HEADS, BATCH * HEADS, 1.0f);
    softmax_rows_k<<<dim3((unsigned)(NSIG * TFR)), TB, 0, stream>>>(scoreF, probs[m], TFR, scale);
  }

  // ---- 5. apply: out_r / out_i (batched, B non-transposed) ----
  zero(outHr, (long)NSIG * TFR * HD);
  zero(outHi, (long)NSIG * TFR * HD);
  const unsigned short* vr = qkvrb + 2 * DMODEL;
  const unsigned short* vi = qkvib + 2 * DMODEL;
  struct Ap { int m; const unsigned short* v; float* C; float a; };
  Ap aps[8] = {
    {0, vr, outHr, +1.f}, {1, vi, outHr, +1.f}, {2, vi, outHr, -1.f}, {3, vr, outHr, -1.f},
    {0, vi, outHi, +1.f}, {1, vr, outHi, +1.f}, {2, vr, outHi, +1.f}, {3, vi, outHi, -1.f}
  };
  for (int x = 0; x < 8; ++x) {
    gemm(false, probs[aps[x].m], aps[x].v, aps[x].C, TFR, HD, TFR,
         TFR, 3 * DMODEL, HD,
         (long)HEADS * TFR * TFR, (long)TFR * TFR, sQKb, HD,
         (long)HEADS * TFR * HD, (long)TFR * HD, HEADS, BATCH * HEADS, aps[x].a);
  }

  // ---- 6. merge heads, output projection ----
  merge_heads_k<<<blks(MROWS * DMODEL), TB, 0, stream>>>(outHr, mergeRb);
  merge_heads_k<<<blks(MROWS * DMODEL), TB, 0, stream>>>(outHi, mergeIb);
  zero(attR, MROWS * DMODEL);
  binit(attI, bo, 2.0f, nullptr, 0.f, MROWS, DMODEL);
  gemm(true, mergeRb, WoB, attR, (int)MROWS, DMODEL, DMODEL, DMODEL, DMODEL, DMODEL, 0,0,0,0,0,0, 1, 1, 1.0f);
  gemm(true, mergeIb, WoB, attI, (int)MROWS, DMODEL, DMODEL, DMODEL, DMODEL, DMODEL, 0,0,0,0,0,0, 1, 1, 1.0f);

  // ---- 7. LayerNorm 1 ----
  layernorm_rows_k<<<dim3((unsigned)MROWS), TB, 0, stream>>>(attR, n1gr, n1br, nullptr, normRb, DMODEL);
  layernorm_rows_k<<<dim3((unsigned)MROWS), TB, 0, stream>>>(attI, n1gi, n1bi, nullptr, normIb, DMODEL);

  // ---- 8. FFN layer 1 (+complex bias), ReLU ----
  binit(hrF, l1br, 1.f, l1bi, -1.f, MROWS, DFF);
  binit(hiF, l1br, 1.f, l1bi, +1.f, MROWS, DFF);
  gemm(true, normRb, l1wrB, hrF, (int)MROWS, DFF, DMODEL, DMODEL, DMODEL, DFF, 0,0,0,0,0,0, 1, 1, +1.0f);
  gemm(true, normIb, l1wiB, hrF, (int)MROWS, DFF, DMODEL, DMODEL, DMODEL, DFF, 0,0,0,0,0,0, 1, 1, -1.0f);
  gemm(true, normIb, l1wrB, hiF, (int)MROWS, DFF, DMODEL, DMODEL, DMODEL, DFF, 0,0,0,0,0,0, 1, 1, +1.0f);
  gemm(true, normRb, l1wiB, hiF, (int)MROWS, DFF, DMODEL, DMODEL, DMODEL, DFF, 0,0,0,0,0,0, 1, 1, +1.0f);
  relu_bf16_k<<<blks(MROWS * DFF), TB, 0, stream>>>(hrF, hrB, MROWS * DFF);
  relu_bf16_k<<<blks(MROWS * DFF), TB, 0, stream>>>(hiF, hiB, MROWS * DFF);

  // ---- 9. FFN layer 2, LayerNorm 2 ----
  binit(l2rF, l2br, 1.f, l2bi, -1.f, MROWS, DMODEL);
  binit(l2iF, l2br, 1.f, l2bi, +1.f, MROWS, DMODEL);
  gemm(true, hrB, l2wrB, l2rF, (int)MROWS, DMODEL, DFF, DFF, DFF, DMODEL, 0,0,0,0,0,0, 1, 1, +1.0f);
  gemm(true, hiB, l2wiB, l2rF, (int)MROWS, DMODEL, DFF, DFF, DFF, DMODEL, 0,0,0,0,0,0, 1, 1, -1.0f);
  gemm(true, hiB, l2wrB, l2iF, (int)MROWS, DMODEL, DFF, DFF, DFF, DMODEL, 0,0,0,0,0,0, 1, 1, +1.0f);
  gemm(true, hrB, l2wiB, l2iF, (int)MROWS, DMODEL, DFF, DFF, DFF, DMODEL, 0,0,0,0,0,0, 1, 1, +1.0f);
  layernorm_rows_k<<<dim3((unsigned)MROWS), TB, 0, stream>>>(l2rF, n2gr, n2br, frF, nullptr, DMODEL);
  layernorm_rows_k<<<dim3((unsigned)MROWS), TB, 0, stream>>>(l2iF, n2gi, n2bi, fiF, nullptr, DMODEL);

  // ---- 10. sigmoid mask * spec; emit spec output + bf16 for iSTFT ----
  mask_apply_k<<<blks(SROWS * BINS), TB, 0, stream>>>(frF, fiF, specR, specI,
      c2wr, c2br, c2wi, c2bi, s2Rb, s2Ib, specOut);

  // ---- 11. iSTFT as DFT GEMM + windowed overlap-add ----
  zero(framesOut, SROWS * NFFT);
  gemm(true, s2Rb, icosW, framesOut, (int)SROWS, NFFT, BINS, BINS, BINS, NFFT, 0,0,0,0,0,0, 1, 1, 1.0f);
  gemm(true, s2Ib, isinW, framesOut, (int)SROWS, NFFT, BINS, BINS, BINS, NFFT, 0,0,0,0,0,0, 1, 1, 1.0f);
  zero(outAcc, (long)NSIG * OUTLEN);
  zero(wsqBuf, (long)OUTLEN);
  wsq_k<<<blks((long)TFR * NFFT), TB, 0, stream>>>(window, wsqBuf);
  ola_k<<<blks(SROWS * NFFT), TB, 0, stream>>>(framesOut, window, outAcc);
  finalize_k<<<blks(ESTIM_N), TB, 0, stream>>>(outAcc, wsqBuf, estimOut);
}
